// MultiLayerFastWeightODE_84859963834879
// MI455X (gfx1250) — matrix operand order repaired and checked by reading, not compiled, not run on hardware
//
#include <hip/hip_runtime.h>

// ---------------------------------------------------------------------------
// MultiLayerFastWeightODE for MI455X (gfx1250, wave32, WMMA).
// - One-time prepass converts all weights to f16 transposed (N x K) so WMMA
//   B-fragments are contiguous 16-half runs loaded directly from global/L2.
// - GEMM software-pipelines: double-buffered LDS A-tile filled by async
//   global->LDS DMA (s_wait_asynccnt 1 overlap), next-chunk B fragment
//   prefetched from global before the current chunk's 4 back-to-back WMMAs.
// - All dense GEMMs: v_wmma_f32_16x16x32_f16, f32 accumulate.
// - State algebra (RK4 3/8 rule) and tanh outer products use float4 I/O.
// ---------------------------------------------------------------------------

#define B_    64
#define L_    32
#define H_    8
#define D_    64
#define HID_  512
#define FW_   32768          // H*D*D
#define ZROW_ 65536          // 2*FW per batch row
#define INCH_ 128
#define FFD_  2048
#define OUT_  128
#define KVBN_ 1032           // 2*HID + H

typedef __attribute__((ext_vector_type(16))) _Float16 v16h;
typedef __attribute__((ext_vector_type(8)))  _Float16 v8h;
typedef __attribute__((ext_vector_type(8)))  float    v8f;
typedef __attribute__((ext_vector_type(4)))  float    v4f;

union HFrag { v16h v; v8h h[2]; _Float16 s[16]; };

#if defined(__has_builtin)
#if __has_builtin(__builtin_amdgcn_global_load_async_to_lds_b128)
#define USE_ASYNC_LDS 1
typedef __attribute__((ext_vector_type(4))) int v4i_;
#endif
#endif

__device__ __forceinline__ void async_copy16(_Float16* dst, const _Float16* src)
{
#if USE_ASYNC_LDS
    __builtin_amdgcn_global_load_async_to_lds_b128(
        (__attribute__((address_space(1))) v4i_*)const_cast<_Float16*>(src),
        (__attribute__((address_space(3))) v4i_*)dst, 0, 0);
#else
    *(v8h*)dst = *(const v8h*)src;
#endif
}

#if USE_ASYNC_LDS
#if __has_builtin(__builtin_amdgcn_s_wait_asynccnt)
#define WAIT_ASYNC(n) __builtin_amdgcn_s_wait_asynccnt(n)
#else
#define WAIT_ASYNC(n) asm volatile("s_wait_asynccnt " #n ::: "memory")
#endif
#else
#define WAIT_ASYNC(n) ((void)0)
#endif

// ---------------------------------------------------------------------------
// Weight prepass: W (K x N, f32 row-major) -> Wt (N x K, f16 col-major).
// ---------------------------------------------------------------------------
__global__ __launch_bounds__(256)
void wtrans_kernel(const float* __restrict__ W, _Float16* __restrict__ Wt,
                   int K, int N)
{
    int idx = blockIdx.x * 256 + threadIdx.x;
    const int total = K * N;
    const int stride = gridDim.x * 256;
    for (; idx < total; idx += stride) {
        const int k = idx / N, n = idx - k * N;     // coalesced read over n
        Wt[(size_t)n * K + k] = (_Float16)W[idx];
    }
}

// f32 -> f16 copy (same layout), for logsig / initial.
__global__ __launch_bounds__(256)
void f32_to_f16_kernel(const float* __restrict__ x, _Float16* __restrict__ y,
                       int n)
{
    const int i = blockIdx.x * 256 + threadIdx.x;
    if (i < n) y[i] = (_Float16)x[i];
}

// ---------------------------------------------------------------------------
// GEMM: C(64,N) = A(64,K)f16 @ Wt(N,K)f16^T + bias [,ReLU][,+res]; optional
// f16 mirror of C. Block computes a 64x128 tile with 8 wave32.
// Two-stage pipeline over K-chunks of 32. flags: bit0 = ReLU, bit1 = res add.
// ---------------------------------------------------------------------------
__global__ __launch_bounds__(256)
void gemm_f16_wmma_kernel(const _Float16* __restrict__ A, int lda,
                          const _Float16* __restrict__ Wt,   // (N,K) col-major
                          const float* __restrict__ bias,
                          const float* __restrict__ res, int ldr,
                          float* __restrict__ C, int ldc,
                          _Float16* __restrict__ Ch,
                          int K, int N, int flags)
{
    __shared__ __attribute__((aligned(16))) _Float16 aS[2][64 * 32];  // 8 KB

    const int tid  = threadIdx.x;
    const int lane = tid & 31;
    const int wave = tid >> 5;
    const int n0   = blockIdx.x * 128;

    // Output / B-source column for this lane (clamped in N-tail blocks).
    const int colw = n0 + wave * 16 + (lane & 15);
    const int bcol = colw < N ? colw : N - 1;
    // B fragment halves i: K = kc + (lane/16)*16 + i  -> contiguous in Wt.
    const _Float16* __restrict__ Bp = Wt + (size_t)bcol * K + ((lane >> 4) << 4);

    // A staging: thread t copies halves [q*8, q*8+8) of row r = t/4.
    const int r = tid >> 2, q = tid & 3;
    const _Float16* __restrict__ Ap = A + (size_t)r * lda + q * 8;
    _Float16* __restrict__ aDst[2] = { aS[0] + r * 32 + q * 8,
                                       aS[1] + r * 32 + q * 8 };

    const int nch = K >> 5;                    // K is a multiple of 32
    v8f acc[4] = { {}, {}, {}, {} };

    // ---- prologue: stage chunk 0, prefetch B fragment for chunk 0 ----
    async_copy16(aDst[0], Ap);
    HFrag bf;
    bf.h[0] = *(const v8h*)(Bp);
    bf.h[1] = *(const v8h*)(Bp + 8);

    for (int i = 0; i < nch; ++i) {
        const int kc = i << 5;
        const _Float16* bufCur = aS[i & 1];

        // issue next chunk's DMA; wait only for the older (in-order) copy
        if (i + 1 < nch) {
            async_copy16(aDst[(i + 1) & 1], Ap + kc + 32);
            WAIT_ASYNC(1);
        } else {
            WAIT_ASYNC(0);
        }
        __syncthreads();                       // chunk i visible to all waves

        // prefetch next B fragment so its loads overlap the WMMAs below
        HFrag bfn;
        if (i + 1 < nch) {
            bfn.h[0] = *(const v8h*)(Bp + kc + 32);
            bfn.h[1] = *(const v8h*)(Bp + kc + 40);
        }

        // gather all 4 A fragments first (batched ds_load_b128, one wait)
        // A fragment halves i: K = (lane/16)*16 + i
        HFrag af[4];
        #pragma unroll
        for (int m = 0; m < 4; ++m) {
            const _Float16* as = bufCur + (m * 16 + (lane & 15)) * 32
                                        + ((lane >> 4) << 4);
            af[m].h[0] = *(const v8h*)(as);
            af[m].h[1] = *(const v8h*)(as + 8);
        }
        // 4 back-to-back WMMAs (independent accumulators)
        #pragma unroll
        for (int m = 0; m < 4; ++m)
            acc[m] = __builtin_amdgcn_wmma_f32_16x16x32_f16(
                false, af[m].v, false, bf.v, (short)0, acc[m], false, false);

        if (i + 1 < nch) bf = bfn;
        __syncthreads();           // WAR: chunk i+2 overwrites bufCur
    }

    // C/D layout: VGPR rr -> row = m*16 + rr + 8*(lane/16); col = lane%16.
    if (colw < N) {
        const float bv = bias ? bias[colw] : 0.0f;
        #pragma unroll
        for (int m = 0; m < 4; ++m) {
            #pragma unroll
            for (int rr = 0; rr < 8; ++rr) {
                const int row = m * 16 + rr + ((lane >> 4) << 3);
                float v = acc[m][rr] + bv;
                if (flags & 1) v = v > 0.0f ? v : 0.0f;
                if (flags & 2) v += res[(size_t)row * ldr + colw];
                C[(size_t)row * ldc + colw] = v;
                if (Ch) Ch[(size_t)row * ldc + colw] = (_Float16)v;
            }
        }
    }
}

// ---------------------------------------------------------------------------
// LayerNorm over 512 cols, one block per row; writes f32 + f16 mirrors.
// ---------------------------------------------------------------------------
__global__ __launch_bounds__(256)
void layernorm_kernel(const float* __restrict__ x, const float* __restrict__ g,
                      const float* __restrict__ b, float* __restrict__ y,
                      _Float16* __restrict__ yh)
{
    __shared__ float red[256];
    __shared__ float s_mu, s_rstd;
    const int row = blockIdx.x, tid = threadIdx.x;
    const float v0 = x[row * HID_ + tid];
    const float v1 = x[row * HID_ + 256 + tid];

    red[tid] = v0 + v1;
    __syncthreads();
    for (int o = 128; o > 0; o >>= 1) {
        if (tid < o) red[tid] += red[tid + o];
        __syncthreads();
    }
    if (tid == 0) s_mu = red[0] * (1.0f / HID_);
    __syncthreads();
    const float mu = s_mu;
    const float d0 = v0 - mu, d1 = v1 - mu;

    red[tid] = d0 * d0 + d1 * d1;
    __syncthreads();
    for (int o = 128; o > 0; o >>= 1) {
        if (tid < o) red[tid] += red[tid + o];
        __syncthreads();
    }
    if (tid == 0) s_rstd = rsqrtf(red[0] * (1.0f / HID_) + 1e-5f);
    __syncthreads();
    const float rs = s_rstd;

    const float o0 = d0 * rs * g[tid] + b[tid];
    const float o1 = d1 * rs * g[tid + 256] + b[tid + 256];
    y[row * HID_ + tid]        = o0;
    y[row * HID_ + 256 + tid]  = o1;
    yh[row * HID_ + tid]       = (_Float16)o0;
    yh[row * HID_ + 256 + tid] = (_Float16)o1;
}

// ---------------------------------------------------------------------------
// Per-head softmax over D=64 (one block per (b,h)); optional sigmoid beta.
// ---------------------------------------------------------------------------
__global__ __launch_bounds__(64)
void head_softmax_kernel(const float* __restrict__ in, int stride,
                         float* __restrict__ out,
                         const float* __restrict__ beta_src,
                         float* __restrict__ beta_out)
{
    __shared__ float red[64];
    const int bh = blockIdx.x, b = bh >> 3, h = bh & 7, j = threadIdx.x;
    const float v = in[(size_t)b * stride + h * 64 + j];

    red[j] = v;
    __syncthreads();
    for (int o = 32; o > 0; o >>= 1) {
        if (j < o) red[j] = fmaxf(red[j], red[j + o]);
        __syncthreads();
    }
    const float m = red[0];
    __syncthreads();
    const float e = __expf(v - m);
    red[j] = e;
    __syncthreads();
    for (int o = 32; o > 0; o >>= 1) {
        if (j < o) red[j] += red[j + o];
        __syncthreads();
    }
    out[b * HID_ + h * 64 + j] = e / red[0];

    if (beta_out != nullptr && j == 0)
        beta_out[b * H_ + h] =
            1.0f / (1.0f + __expf(-beta_src[(size_t)b * stride + h]));
}

// ---------------------------------------------------------------------------
// Delta-rule update: v' = v - W k (if W), out = tanh(beta * v' k^T).
// One block per (b,h); float4 output stores.
// ---------------------------------------------------------------------------
__global__ __launch_bounds__(256)
void delta_tanh_kernel(const float* __restrict__ Wbase, long wrow,
                       const float* __restrict__ ksoft,
                       const float* __restrict__ vsrc, int vstride,
                       const float* __restrict__ beta,
                       float* __restrict__ outbase, long orow)
{
    __shared__ float kS[64], vS[64], red[256];
    const int bh = blockIdx.x, b = bh >> 3, h = bh & 7, tid = threadIdx.x;
    if (tid < 64) {
        kS[tid] = ksoft[b * HID_ + h * 64 + tid];
        vS[tid] = vsrc[(size_t)b * vstride + h * 64 + tid];
    }
    __syncthreads();

    if (Wbase != nullptr) {
        const int d = tid >> 2, p = tid & 3;
        const float* Wr = Wbase + (size_t)b * wrow + h * 4096 + d * 64 + p * 16;
        float s = 0.0f;
        #pragma unroll
        for (int e = 0; e < 16; ++e) s += Wr[e] * kS[p * 16 + e];
        red[tid] = s;
        __syncthreads();
        if (tid < 64)
            vS[tid] -= red[tid * 4] + red[tid * 4 + 1] +
                       red[tid * 4 + 2] + red[tid * 4 + 3];
        __syncthreads();
    }

    const float bt = beta[b * H_ + h];
    float* o = outbase + (size_t)b * orow + h * 4096;
    #pragma unroll
    for (int i4 = tid; i4 < 1024; i4 += 256) {       // 64 rows x 16 quads
        const int d = i4 >> 4;
        const int e0 = (i4 & 15) << 2;
        const float bv = bt * vS[d];
        v4f o4;
        o4[0] = tanhf(bv * kS[e0 + 0]);
        o4[1] = tanhf(bv * kS[e0 + 1]);
        o4[2] = tanhf(bv * kS[e0 + 2]);
        o4[3] = tanhf(bv * kS[e0 + 3]);
        *(v4f*)(o + d * 64 + e0) = o4;
    }
}

// ---------------------------------------------------------------------------
// Readout: o[b, h*64+d] = sum_e W[b,h,d,e] * q[b,h*64+e]; optional f16 mirror.
// ---------------------------------------------------------------------------
__global__ __launch_bounds__(256)
void readout_kernel(const float* __restrict__ Wbase, long wrow,
                    const float* __restrict__ q, float* __restrict__ o,
                    _Float16* __restrict__ oh)
{
    __shared__ float qS[64], red[256];
    const int bh = blockIdx.x, b = bh >> 3, h = bh & 7, tid = threadIdx.x;
    if (tid < 64) qS[tid] = q[b * HID_ + h * 64 + tid];
    __syncthreads();
    const int d = tid >> 2, p = tid & 3;
    const float* Wr = Wbase + (size_t)b * wrow + h * 4096 + d * 64 + p * 16;
    float s = 0.0f;
    #pragma unroll
    for (int e = 0; e < 16; ++e) s += Wr[e] * qS[p * 16 + e];
    red[tid] = s;
    __syncthreads();
    if (tid < 64) {
        const float v = red[tid * 4] + red[tid * 4 + 1] +
                        red[tid * 4 + 2] + red[tid * 4 + 3];
        o[b * HID_ + h * 64 + tid] = v;
        if (oh) oh[b * HID_ + h * 64 + tid] = (_Float16)v;
    }
}

// ---------------------------------------------------------------------------
// dst = base + a1*p1 + a2*p2 + a3*p3 + a4*p4, float4-wide, exact-fit grid.
// ---------------------------------------------------------------------------
__global__ __launch_bounds__(256)
void combine4_kernel(v4f* __restrict__ dst, const v4f* __restrict__ base,
                     const v4f* __restrict__ p1, float a1,
                     const v4f* __restrict__ p2, float a2,
                     const v4f* __restrict__ p3, float a3,
                     const v4f* __restrict__ p4, float a4)
{
    const int i = blockIdx.x * 256 + threadIdx.x;
    dst[i] = base[i] + a1 * p1[i] + a2 * p2[i] + a3 * p3[i] + a4 * p4[i];
}

// Zero z[b, FW:2FW], float4-wide. Grid covers B*FW/4 elements.
__global__ __launch_bounds__(256)
void zero_upper_kernel(float* __restrict__ z)
{
    const int i = blockIdx.x * 256 + threadIdx.x;   // < B*FW/4
    const int b = i >> 13;                          // FW/4 = 8192
    const int c4 = i & 8191;
    *((v4f*)(z + (size_t)b * ZROW_ + FW_) + c4) = (v4f){0.f, 0.f, 0.f, 0.f};
}

// ---------------------------------------------------------------------------
// Host orchestration
// ---------------------------------------------------------------------------
namespace {

struct Ctx {
    // f32 inputs
    const float *initial, *logsig;
    const float *init_b, *proj_b, *un1_ln_g, *un1_ln_b, *un1_kvb_b;
    const float *un2_ln_g, *un2_ln_b, *un2_kvb_b;
    const float *mq_b, *q_b, *op_b;
    const float *mff_ln_g, *mff_ln_b, *mff_b1, *mff_b2;
    const float *ff_ln_g, *ff_ln_b, *ff_b1, *ff_b2, *fin_b;
    // f16 transposed weights (workspace)
    _Float16 *init_Wt, *proj_Wt, *kvb1_Wt, *kvb2_Wt, *mq_Wt, *q_Wt, *op_Wt;
    _Float16 *mff1_Wt, *mff2_Wt, *ff1_Wt, *ff2_Wt, *fin_Wt;
    // f16 activations (workspace)
    _Float16 *logsig16, *initial16, *t128h, *hn16, *o1n16, *fft16, *o2n16;
    _Float16 *o1h, *o1f16;
    // f32 workspace
    float *z, *ztmp, *k1, *k2, *k3, *k4;
    float *h, *hn, *kvb, *ksoft, *beta, *qb, *qsoft;
    float *o1, *o1n, *fft, *o1f, *o2n, *kvb2, *ksoft2, *beta2, *t128;
    hipStream_t s;
};

inline void gemm(const Ctx& c, const _Float16* A, int lda, const _Float16* Wt,
                 const float* bias, const float* res, int ldr,
                 float* C, int ldc, _Float16* Ch, int K, int N, int flags)
{
    dim3 grid((N + 127) / 128);
    gemm_f16_wmma_kernel<<<grid, 256, 0, c.s>>>(A, lda, Wt, bias, res, ldr,
                                                C, ldc, Ch, K, N, flags);
}

inline void ln(const Ctx& c, const float* x, const float* g, const float* b,
               float* y, _Float16* yh)
{
    layernorm_kernel<<<B_, 256, 0, c.s>>>(x, g, b, y, yh);
}

inline void hsoftmax(const Ctx& c, const float* in, int stride, float* out,
                     const float* bsrc, float* bout)
{
    head_softmax_kernel<<<B_ * H_, 64, 0, c.s>>>(in, stride, out, bsrc, bout);
}

inline void delta(const Ctx& c, const float* Wb, const float* ks,
                  const float* vs, int vstride, const float* bt, float* out)
{
    delta_tanh_kernel<<<B_ * H_, 256, 0, c.s>>>(Wb, (long)ZROW_, ks, vs,
                                                vstride, bt, out, (long)ZROW_);
}

inline void readout(const Ctx& c, const float* Wb, const float* q, float* o,
                    _Float16* oh)
{
    readout_kernel<<<B_ * H_, 256, 0, c.s>>>(Wb, (long)ZROW_, q, o, oh);
}

inline void comb(const Ctx& c, float* dst, const float* base,
                 const float* p1, float a1, const float* p2, float a2,
                 const float* p3, float a3, const float* p4, float a4)
{
    combine4_kernel<<<(B_ * ZROW_) / 4 / 256, 256, 0, c.s>>>(
        (v4f*)dst, (const v4f*)base, (const v4f*)p1, a1, (const v4f*)p2, a2,
        (const v4f*)p3, a3, (const v4f*)p4, a4);
}

inline void wtrans(const Ctx& c, const float* W, _Float16* Wt, int K, int N)
{
    wtrans_kernel<<<(K * N) / 256, 256, 0, c.s>>>(W, Wt, K, N);
}

// One vector-field evaluation: kout = vf(x16, zin)
void vf_eval(const Ctx& c, const float* zin, const _Float16* x16, int lda,
             float* kout)
{
    // layer 1
    gemm(c, x16, lda, c.proj_Wt, c.proj_b, nullptr, 0, c.h, HID_, nullptr,
         INCH_, HID_, 0);
    ln(c, c.h, c.un1_ln_g, c.un1_ln_b, c.hn, c.hn16);
    gemm(c, c.hn16, HID_, c.kvb1_Wt, c.un1_kvb_b, nullptr, 0, c.kvb, KVBN_,
         nullptr, HID_, KVBN_, 0);
    hsoftmax(c, c.kvb, KVBN_, c.ksoft, c.kvb + 2 * HID_, c.beta);
    delta(c, zin, c.ksoft, c.kvb + HID_, KVBN_, c.beta, kout);

    // layer-1 readout + FF
    gemm(c, c.hn16, HID_, c.mq_Wt, c.mq_b, nullptr, 0, c.qb, HID_, nullptr,
         HID_, HID_, 0);
    hsoftmax(c, c.qb, HID_, c.qsoft, nullptr, nullptr);
    readout(c, zin, c.qsoft, c.o1, nullptr);
    ln(c, c.o1, c.mff_ln_g, c.mff_ln_b, c.o1n, c.o1n16);
    gemm(c, c.o1n16, HID_, c.mff1_Wt, c.mff_b1, nullptr, 0, c.fft, FFD_,
         c.fft16, HID_, FFD_, 1);
    gemm(c, c.fft16, FFD_, c.mff2_Wt, c.mff_b2, c.o1, HID_, c.o1f, HID_,
         nullptr, FFD_, HID_, 2);

    // layer 2
    ln(c, c.o1f, c.un2_ln_g, c.un2_ln_b, c.o2n, c.o2n16);
    gemm(c, c.o2n16, HID_, c.kvb2_Wt, c.un2_kvb_b, nullptr, 0, c.kvb2, KVBN_,
         nullptr, HID_, KVBN_, 0);
    hsoftmax(c, c.kvb2, KVBN_, c.ksoft2, c.kvb2 + 2 * HID_, c.beta2);
    delta(c, zin + FW_, c.ksoft2, c.kvb2 + HID_, KVBN_, c.beta2, kout + FW_);
}

} // namespace

extern "C" void kernel_launch(void* const* d_in, const int* in_sizes, int n_in,
                              void* d_out, int out_size, void* d_ws,
                              size_t ws_size, hipStream_t stream)
{
    (void)in_sizes; (void)n_in; (void)out_size; (void)ws_size;
    Ctx c;
    c.s = stream;
    const float* init_W    = (const float*)d_in[2];
    const float* proj_W    = (const float*)d_in[4];
    const float* un1_kvb_W = (const float*)d_in[8];
    const float* un2_kvb_W = (const float*)d_in[12];
    const float* mq_W      = (const float*)d_in[14];
    const float* q_W       = (const float*)d_in[16];
    const float* op_W      = (const float*)d_in[18];
    const float* mff_W1    = (const float*)d_in[22];
    const float* mff_W2    = (const float*)d_in[24];
    const float* ff_W1     = (const float*)d_in[28];
    const float* ff_W2     = (const float*)d_in[30];
    const float* fin_W     = (const float*)d_in[32];

    c.initial   = (const float*)d_in[0];
    c.logsig    = (const float*)d_in[1];
    c.init_b    = (const float*)d_in[3];
    c.proj_b    = (const float*)d_in[5];
    c.un1_ln_g  = (const float*)d_in[6];
    c.un1_ln_b  = (const float*)d_in[7];
    c.un1_kvb_b = (const float*)d_in[9];
    c.un2_ln_g  = (const float*)d_in[10];
    c.un2_ln_b  = (const float*)d_in[11];
    c.un2_kvb_b = (const float*)d_in[13];
    c.mq_b      = (const float*)d_in[15];
    c.q_b       = (const float*)d_in[17];
    c.op_b      = (const float*)d_in[19];
    c.mff_ln_g  = (const float*)d_in[20];
    c.mff_ln_b  = (const float*)d_in[21];
    c.mff_b1    = (const float*)d_in[23];
    c.mff_b2    = (const float*)d_in[25];
    c.ff_ln_g   = (const float*)d_in[26];
    c.ff_ln_b   = (const float*)d_in[27];
    c.ff_b1     = (const float*)d_in[29];
    c.ff_b2     = (const float*)d_in[31];
    c.fin_b     = (const float*)d_in[33];

    // ---------------- workspace layout ----------------
    const size_t NZ = (size_t)B_ * ZROW_;
    float* ws = (float*)d_ws;
    c.z    = ws;
    c.ztmp = c.z + NZ;
    c.k1   = c.ztmp + NZ;
    c.k2   = c.k1 + NZ;
    c.k3   = c.k2 + NZ;
    c.k4   = c.k3 + NZ;
    float* p = c.k4 + NZ;
    c.h      = p; p += B_ * HID_;
    c.hn     = p; p += B_ * HID_;
    c.kvb    = p; p += B_ * KVBN_;
    c.ksoft  = p; p += B_ * HID_;
    c.beta   = p; p += B_ * H_;
    c.qb     = p; p += B_ * HID_;
    c.qsoft  = p; p += B_ * HID_;
    c.o1     = p; p += B_ * HID_;
    c.o1n    = p; p += B_ * HID_;
    c.fft    = p; p += B_ * FFD_;
    c.o1f    = p; p += B_ * HID_;
    c.o2n    = p; p += B_ * HID_;
    c.kvb2   = p; p += B_ * KVBN_;
    c.ksoft2 = p; p += B_ * HID_;
    c.beta2  = p; p += B_ * H_;
    c.t128   = p; p += B_ * INCH_;

    _Float16* hp = (_Float16*)p;                 // 16B-aligned (sizes are /4)
    c.init_Wt = hp; hp += INCH_ * INCH_;
    c.proj_Wt = hp; hp += INCH_ * HID_;
    c.kvb1_Wt = hp; hp += HID_ * KVBN_;
    c.kvb2_Wt = hp; hp += HID_ * KVBN_;
    c.mq_Wt   = hp; hp += HID_ * HID_;
    c.q_Wt    = hp; hp += HID_ * HID_;
    c.op_Wt   = hp; hp += HID_ * HID_;
    c.mff1_Wt = hp; hp += HID_ * FFD_;
    c.mff2_Wt = hp; hp += FFD_ * HID_;
    c.ff1_Wt  = hp; hp += HID_ * FFD_;
    c.ff2_Wt  = hp; hp += FFD_ * HID_;
    c.fin_Wt  = hp; hp += HID_ * OUT_;
    c.logsig16  = hp; hp += B_ * L_ * INCH_;
    c.initial16 = hp; hp += B_ * INCH_;
    c.t128h     = hp; hp += B_ * INCH_;
    c.hn16      = hp; hp += B_ * HID_;
    c.o1n16     = hp; hp += B_ * HID_;
    c.fft16     = hp; hp += B_ * FFD_;
    c.o2n16     = hp; hp += B_ * HID_;
    c.o1h       = hp; hp += B_ * HID_;
    c.o1f16     = hp; hp += B_ * HID_;

    float* out = (float*)d_out;

    // ---------------- prepass: f16 transposed weights + f16 inputs --------
    wtrans(c, init_W, c.init_Wt, INCH_, INCH_);
    wtrans(c, proj_W, c.proj_Wt, INCH_, HID_);
    wtrans(c, un1_kvb_W, c.kvb1_Wt, HID_, KVBN_);
    wtrans(c, un2_kvb_W, c.kvb2_Wt, HID_, KVBN_);
    wtrans(c, mq_W, c.mq_Wt, HID_, HID_);
    wtrans(c, q_W, c.q_Wt, HID_, HID_);
    wtrans(c, op_W, c.op_Wt, HID_, HID_);
    wtrans(c, mff_W1, c.mff1_Wt, HID_, FFD_);
    wtrans(c, mff_W2, c.mff2_Wt, FFD_, HID_);
    wtrans(c, ff_W1, c.ff1_Wt, HID_, FFD_);
    wtrans(c, ff_W2, c.ff2_Wt, FFD_, HID_);
    wtrans(c, fin_W, c.fin_Wt, HID_, OUT_);
    f32_to_f16_kernel<<<(B_ * L_ * INCH_) / 256, 256, 0, stream>>>(
        c.logsig, c.logsig16, B_ * L_ * INCH_);
    f32_to_f16_kernel<<<(B_ * INCH_) / 256, 256, 0, stream>>>(
        c.initial, c.initial16, B_ * INCH_);

    // ---- init: z1 = delta_update(proj(init_net(initial)), W=None); z2=0 --
    gemm(c, c.initial16, INCH_, c.init_Wt, c.init_b, nullptr, 0, c.t128, INCH_,
         c.t128h, INCH_, INCH_, 0);
    gemm(c, c.t128h, INCH_, c.proj_Wt, c.proj_b, nullptr, 0, c.h, HID_,
         nullptr, INCH_, HID_, 0);
    ln(c, c.h, c.un1_ln_g, c.un1_ln_b, c.hn, c.hn16);
    gemm(c, c.hn16, HID_, c.kvb1_Wt, c.un1_kvb_b, nullptr, 0, c.kvb, KVBN_,
         nullptr, HID_, KVBN_, 0);
    hsoftmax(c, c.kvb, KVBN_, c.ksoft, c.kvb + 2 * HID_, c.beta);
    delta(c, nullptr, c.ksoft, c.kvb + HID_, KVBN_, c.beta, c.z);
    zero_upper_kernel<<<(B_ * FW_) / 4 / 256, 256, 0, stream>>>(c.z);

    // ---------------- RK4 (3/8 rule), dt = 1 ----------------
    const float third = 1.0f / 3.0f;
    for (int t = 0; t < L_; ++t) {
        const _Float16* xt = c.logsig16 + t * INCH_;
        const int tn = (t + 1 < L_) ? (t + 1) : (L_ - 1);
        const _Float16* xn = c.logsig16 + tn * INCH_;
        const int lda = L_ * INCH_;

        vf_eval(c, c.z, xt, lda, c.k1);
        comb(c, c.ztmp, c.z, c.k1, third, c.k1, 0.f, c.k1, 0.f, c.k1, 0.f);
        vf_eval(c, c.ztmp, xt, lda, c.k2);
        comb(c, c.ztmp, c.z, c.k2, 1.f, c.k1, -third, c.k1, 0.f, c.k1, 0.f);
        vf_eval(c, c.ztmp, xt, lda, c.k3);
        comb(c, c.ztmp, c.z, c.k1, 1.f, c.k2, -1.f, c.k3, 1.f, c.k1, 0.f);
        vf_eval(c, c.ztmp, xn, lda, c.k4);
        comb(c, c.z, c.z, c.k1, 0.125f, c.k2, 0.375f, c.k3, 0.375f,
             c.k4, 0.125f);
    }

    // ---------------- final readout ----------------
    const _Float16* xq = c.logsig16 + (L_ - 1) * INCH_;
    gemm(c, xq, L_ * INCH_, c.proj_Wt, c.proj_b, nullptr, 0, c.h, HID_,
         nullptr, INCH_, HID_, 0);
    ln(c, c.h, c.un1_ln_g, c.un1_ln_b, c.hn, c.hn16);
    gemm(c, c.hn16, HID_, c.mq_Wt, c.mq_b, nullptr, 0, c.qb, HID_, nullptr,
         HID_, HID_, 0);
    hsoftmax(c, c.qb, HID_, c.qsoft, nullptr, nullptr);
    readout(c, c.z, c.qsoft, c.o1, nullptr);
    ln(c, c.o1, c.mff_ln_g, c.mff_ln_b, c.o1n, c.o1n16);
    gemm(c, c.o1n16, HID_, c.mff1_Wt, c.mff_b1, nullptr, 0, c.fft, FFD_,
         c.fft16, HID_, FFD_, 1);
    gemm(c, c.fft16, FFD_, c.mff2_Wt, c.mff_b2, c.o1, HID_, c.o1f, HID_,
         nullptr, FFD_, HID_, 2);

    ln(c, c.o1f, c.un2_ln_g, c.un2_ln_b, c.o2n, c.o2n16);
    gemm(c, c.o2n16, HID_, c.q_Wt, c.q_b, nullptr, 0, c.qb, HID_, nullptr,
         HID_, HID_, 0);
    hsoftmax(c, c.qb, HID_, c.qsoft, nullptr, nullptr);
    readout(c, c.z + FW_, c.qsoft, c.o1, c.o1h);

    gemm(c, c.o1h, HID_, c.op_Wt, c.op_b, nullptr, 0, c.h, HID_, nullptr,
         HID_, HID_, 0);
    ln(c, c.h, c.ff_ln_g, c.ff_ln_b, c.hn, c.hn16);
    gemm(c, c.hn16, HID_, c.ff1_Wt, c.ff_b1, nullptr, 0, c.fft, FFD_,
         c.fft16, HID_, FFD_, 1);
    gemm(c, c.fft16, FFD_, c.ff2_Wt, c.ff_b2, c.h, HID_, c.o1f, HID_,
         c.o1f16, FFD_, HID_, 2);
    gemm(c, c.o1f16, HID_, c.fin_Wt, c.fin_b, nullptr, 0, out, OUT_,
         nullptr, HID_, OUT_, 0);
}
